// GraphCNN_88545045774673
// MI455X (gfx1250) — matrix operand order repaired and checked by reading, not compiled, not run on hardware
//
#include <hip/hip_runtime.h>

// ---------------------------------------------------------------------------
// GraphCNN for MI455X (gfx1250): bf16 WMMA everywhere, single pass over graph
// per layer via the global-max softmax factorization.
// Round 2: aggregation kernel re-tiled (32-row tiles -> 256 WGs for device
// fill) with a double-buffered LDS weight tile (one barrier per K-step,
// next-step graph loads overlap the WMMAs).
// ---------------------------------------------------------------------------

typedef __attribute__((ext_vector_type(8)))  __bf16 bf16x8;
typedef __attribute__((ext_vector_type(16))) __bf16 bf16x16;
typedef __attribute__((ext_vector_type(8)))  float  f32x8;

static constexpr int NN   = 8192;   // nodes
static constexpr int DD   = 256;    // feature dim
static constexpr int ATTH = 64;     // attention hidden
static constexpr int MH   = 54;     // mask hidden
static constexpr int LL   = 3;      // conv layers
static constexpr int OUTD = 64;     // output dim

__device__ __forceinline__ unsigned short f2bf(float f) {
  unsigned u = __float_as_uint(f);
  unsigned r = u + 0x7FFFu + ((u >> 16) & 1u);   // round-to-nearest-even
  return (unsigned short)(r >> 16);
}
__device__ __forceinline__ float bf2f(unsigned short h) {
  return __uint_as_float(((unsigned)h) << 16);
}
__device__ __forceinline__ bf16x16 cat8(bf16x8 lo, bf16x8 hi) {
  return __builtin_shufflevector(lo, hi, 0,1,2,3,4,5,6,7,8,9,10,11,12,13,14,15);
}

// --------------------------- f32 -> bf16 convert ---------------------------
__global__ void cvt_bf16_kernel(const float* __restrict__ in,
                                unsigned short* __restrict__ out, int n) {
  int i = blockIdx.x * blockDim.x + threadIdx.x;
  if (i < n) out[i] = f2bf(in[i]);
}

// ------------------- generic WMMA GEMM: out = act(A @ B + b) ----------------
// A: [M,K] bf16 row-major, B: [K,N] bf16 row-major, bias: f32[N] or null.
// One wave32 per 16x16 output tile. K % 32 == 0, N % 16 == 0.
template <bool RELU, bool OUT_BF16>
__global__ void __launch_bounds__(32)
gemm_wmma_kernel(const unsigned short* __restrict__ A,
                 const unsigned short* __restrict__ B,
                 const float* __restrict__ bias,
                 void* __restrict__ Out, int M, int N, int K) {
  const int lane = threadIdx.x;
  const int n0 = blockIdx.x * 16;
  const int m0 = blockIdx.y * 16;
  const int half = lane >> 4;
  const int l16  = lane & 15;
  const int kb   = half * 8;

  f32x8 acc;
  const float bv = bias ? bias[n0 + l16] : 0.0f;
#pragma unroll
  for (int i = 0; i < 8; ++i) acc[i] = bv;

  const unsigned short* Ap = A + (size_t)(m0 + l16) * K;
  for (int k0 = 0; k0 < K; k0 += 32) {
    bf16x8 alo = *(const bf16x8*)(Ap + k0 + kb);
    bf16x8 ahi = *(const bf16x8*)(Ap + k0 + kb + 16);
    bf16x16 af = cat8(alo, ahi);
    const unsigned short* Bp = B + (size_t)(k0 + kb) * N + n0 + l16;
    bf16x16 bfv;
#pragma unroll
    for (int i = 0; i < 8; ++i) {
      bfv[i] = __builtin_bit_cast(__bf16, Bp[(size_t)i * N]);
      bfv[i + 8] = __builtin_bit_cast(__bf16, Bp[(size_t)(i + 16) * N]);
    }
    acc = __builtin_amdgcn_wmma_f32_16x16x32_bf16(false, af, false, bfv,
                                                  (short)0, acc, false, false);
  }
#pragma unroll
  for (int r = 0; r < 8; ++r) {
    float v = acc[r];
    if (RELU) v = fmaxf(v, 0.0f);
    size_t idx = (size_t)(m0 + r + half * 8) * N + n0 + l16;
    if (OUT_BF16) ((unsigned short*)Out)[idx] = f2bf(v);
    else          ((float*)Out)[idx] = v;
  }
}

// ----------- fused graph aggregation: h_agg = deg * softmax(G,e) @ h --------
// Weights w[i,j] = (g[i,j]>0) ? p[j] : 0, built on the fly into a double-
// buffered LDS bf16 tile; s_i (softmax denom) and deg_i accumulated in
// registers over the full K sweep (one graph pass per layer).
// Block: 256 threads = 8 waves, output tile 32 rows x 256 cols, K = 8192.
// Wave w: row-tile rt = w&1, columns [(w>>1)*64, +64) = 4 WMMA col tiles.
__global__ void __launch_bounds__(256)
agg_wmma_kernel(const float* __restrict__ graph,
                const float* __restrict__ p,
                const unsigned short* __restrict__ hbf,
                unsigned short* __restrict__ haggbf) {
  __shared__ __align__(16) unsigned short Atile[2][32 * 40]; // 80B row stride
  __shared__ float sS[32];
  __shared__ float sDeg[32];

  const int t = threadIdx.x;
  const int rowbase = blockIdx.x * 32;
  const int row = t >> 3;           // 32 rows, 8 threads/row
  const int cg  = (t & 7) * 4;      // 4 cols per thread per K-step
  const int wave = t >> 5;
  const int lane = t & 31;
  const int rt    = wave & 1;            // row tile 0..1
  const int nbase = (wave >> 1) * 64;    // 4 col quarters
  const int half = lane >> 4;
  const int l16  = lane & 15;
  const int kb   = half * 8;
  const int arow = (16 * rt + l16) * 40;

  float s_loc = 0.0f, d_loc = 0.0f;
  f32x8 acc[4];
#pragma unroll
  for (int ct = 0; ct < 4; ++ct)
#pragma unroll
    for (int i = 0; i < 8; ++i) acc[ct][i] = 0.0f;

  const float* gbase = graph + (size_t)(rowbase + row) * NN + cg;

  // software pipeline: preload step 0
  float4 g4 = *(const float4*)(gbase);
  float4 p4 = *(const float4*)(p + cg);
  int buf = 0;

  for (int k0 = 0; k0 < NN; k0 += 32) {
    // ---- write bf16 weight tile (32x32) into LDS buffer `buf` ----
    float gv[4] = {g4.x, g4.y, g4.z, g4.w};
    float pv[4] = {p4.x, p4.y, p4.z, p4.w};
    unsigned pk[2];
#pragma unroll
    for (int i = 0; i < 4; i += 2) {
      float w0 = gv[i]     > 0.0f ? pv[i]     : 0.0f;
      float w1 = gv[i + 1] > 0.0f ? pv[i + 1] : 0.0f;
      s_loc += w0 + w1;
      d_loc += gv[i] + gv[i + 1];
      pk[i >> 1] = (unsigned)f2bf(w0) | ((unsigned)f2bf(w1) << 16);
    }
    unsigned* dst = (unsigned*)&Atile[buf][row * 40 + cg];
    dst[0] = pk[0];
    dst[1] = pk[1];
    __syncthreads();   // single barrier per step (double-buffered tile)

    // ---- preload next step's graph/p while WMMAs run ----
    if (k0 + 32 < NN) {
      g4 = *(const float4*)(gbase + k0 + 32);
      p4 = *(const float4*)(p + k0 + 32 + cg);
      __builtin_prefetch(gbase + k0 + 96, 0, 1);   // keep the stream deep
    }

    // ---- consume: 4 WMMA per wave against L2-resident h ----
    bf16x8 alo = *(const bf16x8*)&Atile[buf][arow + kb];
    bf16x8 ahi = *(const bf16x8*)&Atile[buf][arow + kb + 16];
    bf16x16 af = cat8(alo, ahi);
    const unsigned short* Bp = hbf + (size_t)(k0 + kb) * DD + nbase + l16;
#pragma unroll
    for (int ct = 0; ct < 4; ++ct) {
      bf16x16 bfv;
#pragma unroll
      for (int i = 0; i < 8; ++i) {
        bfv[i] = __builtin_bit_cast(__bf16, Bp[ct * 16 + i * DD]);
        bfv[i + 8] = __builtin_bit_cast(__bf16, Bp[ct * 16 + (i + 16) * DD]);
      }
      acc[ct] = __builtin_amdgcn_wmma_f32_16x16x32_bf16(
          false, af, false, bfv, (short)0, acc[ct], false, false);
    }
    buf ^= 1;
  }

  // reduce s, deg across the 8 threads covering each row (octet in-wave)
  s_loc += __shfl_xor(s_loc, 1, 32);
  s_loc += __shfl_xor(s_loc, 2, 32);
  s_loc += __shfl_xor(s_loc, 4, 32);
  d_loc += __shfl_xor(d_loc, 1, 32);
  d_loc += __shfl_xor(d_loc, 2, 32);
  d_loc += __shfl_xor(d_loc, 4, 32);
  if ((t & 7) == 0) { sS[row] = s_loc; sDeg[row] = d_loc; }
  __syncthreads();

#pragma unroll
  for (int ct = 0; ct < 4; ++ct) {
#pragma unroll
    for (int r = 0; r < 8; ++r) {
      int rl = 16 * rt + r + half * 8;
      float scale = sDeg[rl] / sS[rl];
      haggbf[(size_t)(rowbase + rl) * DD + nbase + ct * 16 + l16] =
          f2bf(acc[ct][r] * scale);
    }
  }
}

// ------------------- adaptive receptive-field mask MLP ----------------------
__global__ void __launch_bounds__(64)
mask_kernel(const float* __restrict__ x, const float* __restrict__ W1,
            const float* __restrict__ b1, const float* __restrict__ W2,
            const float* __restrict__ b2, float* __restrict__ maskb) {
  __shared__ float xs[DD];
  __shared__ float hid[64];
  int n = blockIdx.x, t = threadIdx.x;
  for (int i = t; i < DD; i += 64) xs[i] = x[(size_t)n * DD + i];
  __syncthreads();
  float acc = 0.0f;
  if (t < MH) {
    for (int k = 0; k < DD; ++k) acc += xs[k] * W1[k * MH + t];
    acc = fmaxf(acc + b1[t], 0.0f);
  }
  hid[t] = (t < MH) ? acc : 0.0f;
  __syncthreads();
  if (t == 0) {
    float s = b2[0];
    for (int i = 0; i < MH; ++i) s += hid[i] * W2[i];
    float thr = (float)LL / (1.0f + expf(-s));   // sigmoid * L
    for (int l = 0; l < LL; ++l) {
      float d = (float)l - thr;
      maskb[n * LL + l] = expf(-d * d);          // ALPHA = 1
    }
  }
}

// ---------------------- e = hidden @ W2 + b2 (dot-64) -----------------------
__global__ void __launch_bounds__(256)
edot_kernel(const unsigned short* __restrict__ hid, const float* __restrict__ W2,
            const float* __restrict__ b2, float* __restrict__ e) {
  int node = blockIdx.x * 8 + (threadIdx.x >> 5);
  int lane = threadIdx.x & 31;
  const unsigned short* hp = hid + (size_t)node * ATTH;
  float v = bf2f(hp[lane]) * W2[lane] + bf2f(hp[lane + 32]) * W2[lane + 32];
  v += __shfl_xor(v, 16, 32);
  v += __shfl_xor(v, 8, 32);
  v += __shfl_xor(v, 4, 32);
  v += __shfl_xor(v, 2, 32);
  v += __shfl_xor(v, 1, 32);
  if (lane == 0) e[node] = v + b2[0];
}

__global__ void __launch_bounds__(1024)
maxred_kernel(const float* __restrict__ e, float* __restrict__ out) {
  __shared__ float red[1024];
  int t = threadIdx.x;
  float m = -3.4e38f;
  for (int i = t; i < NN; i += 1024) m = fmaxf(m, e[i]);
  red[t] = m;
  __syncthreads();
  for (int s = 512; s > 0; s >>= 1) {
    if (t < s) red[t] = fmaxf(red[t], red[t + s]);
    __syncthreads();
  }
  if (t == 0) out[0] = red[0];
}

__global__ void expp_kernel(const float* __restrict__ e,
                            const float* __restrict__ mmax,
                            float* __restrict__ p) {
  int i = blockIdx.x * blockDim.x + threadIdx.x;
  if (i < NN) p[i] = expf(e[i] - mmax[0]);
}

// --------------------------- BatchNorm statistics ---------------------------
__global__ void __launch_bounds__(256)
bnstat1_kernel(const float* __restrict__ h2, float* __restrict__ psum,
               float* __restrict__ psq) {
  int c = threadIdx.x;
  int r0 = blockIdx.x * 128;
  float s = 0.0f, q = 0.0f;
  for (int r = r0; r < r0 + 128; ++r) {
    float v = h2[(size_t)r * DD + c];
    s += v;
    q += v * v;
  }
  psum[blockIdx.x * DD + c] = s;
  psq[blockIdx.x * DD + c] = q;
}

__global__ void __launch_bounds__(256)
bnstat2_kernel(const float* __restrict__ psum, const float* __restrict__ psq,
               float* __restrict__ mean, float* __restrict__ var) {
  int c = threadIdx.x;
  float s = 0.0f, q = 0.0f;
  for (int b = 0; b < 64; ++b) {
    s += psum[b * DD + c];
    q += psq[b * DD + c];
  }
  float m = s / (float)NN;
  float v = q / (float)NN - m * m;
  mean[c] = m;
  var[c] = fmaxf(v, 0.0f);
}

__global__ void __launch_bounds__(256)
bnapply_kernel(const float* __restrict__ h2, const float* __restrict__ mean,
               const float* __restrict__ var, const float* __restrict__ gamma,
               const float* __restrict__ beta, const float* __restrict__ maskb,
               int l, unsigned short* __restrict__ h_bf,
               unsigned short* __restrict__ feat_bf) {
  int n = blockIdx.x, d = threadIdx.x;
  float v = h2[(size_t)n * DD + d];
  v = (v - mean[d]) * rsqrtf(var[d] + 1e-5f) * gamma[d] + beta[d];
  v = fmaxf(v, 0.0f);
  h_bf[(size_t)n * DD + d] = f2bf(v);
  feat_bf[(size_t)n * (LL * DD) + l * DD + d] = f2bf(v * maskb[n * LL + l]);
}

// ---------------------------------------------------------------------------
extern "C" void kernel_launch(void* const* d_in, const int* in_sizes, int n_in,
                              void* d_out, int out_size, void* d_ws,
                              size_t ws_size, hipStream_t stream) {
  const float* graph  = (const float*)d_in[0];
  const float* x      = (const float*)d_in[1];
  const float* att_W1 = (const float*)d_in[2];
  const float* att_b1 = (const float*)d_in[3];
  const float* att_W2 = (const float*)d_in[4];
  const float* att_b2 = (const float*)d_in[5];
  const float* mlp_W1 = (const float*)d_in[6];
  const float* mlp_b1 = (const float*)d_in[7];
  const float* mlp_W2 = (const float*)d_in[8];
  const float* mlp_b2 = (const float*)d_in[9];
  const float* bn_g   = (const float*)d_in[10];
  const float* bn_b   = (const float*)d_in[11];
  const float* mk_W1  = (const float*)d_in[12];
  const float* mk_b1  = (const float*)d_in[13];
  const float* mk_W2  = (const float*)d_in[14];
  const float* mk_b2  = (const float*)d_in[15];
  const float* pred_W = (const float*)d_in[16];
  const float* pred_b = (const float*)d_in[17];

  char* ws = (char*)d_ws;
  size_t off = 0;
  auto alloc = [&](size_t bytes) -> char* {
    char* ptr = ws + off;
    off += (bytes + 255) & ~(size_t)255;
    return ptr;
  };
  unsigned short* h_bf     = (unsigned short*)alloc((size_t)NN * DD * 2);
  unsigned short* hagg_bf  = (unsigned short*)alloc((size_t)NN * DD * 2);
  unsigned short* t1_bf    = (unsigned short*)alloc((size_t)NN * DD * 2);
  unsigned short* atth_bf  = (unsigned short*)alloc((size_t)NN * ATTH * 2);
  float*          h2_f32   = (float*)alloc((size_t)NN * DD * 4);
  unsigned short* feat_bf  = (unsigned short*)alloc((size_t)NN * LL * DD * 2);
  float*          e_buf    = (float*)alloc(NN * 4);
  float*          p_buf    = (float*)alloc(NN * 4);
  float*          mmax     = (float*)alloc(256);
  float*          maskbuf  = (float*)alloc(NN * LL * 4);
  float*          psum     = (float*)alloc(64 * DD * 4);
  float*          psq      = (float*)alloc(64 * DD * 4);
  float*          meanb    = (float*)alloc(DD * 4);
  float*          varb     = (float*)alloc(DD * 4);
  unsigned short* attW1_bf = (unsigned short*)alloc((size_t)LL * DD * ATTH * 2);
  unsigned short* mlpW1_bf = (unsigned short*)alloc((size_t)LL * DD * DD * 2);
  unsigned short* mlpW2_bf = (unsigned short*)alloc((size_t)LL * DD * DD * 2);
  unsigned short* predW_bf = (unsigned short*)alloc((size_t)LL * DD * OUTD * 2);

  auto cvt = [&](const float* src, unsigned short* dst, int n) {
    cvt_bf16_kernel<<<(n + 255) / 256, 256, 0, stream>>>(src, dst, n);
  };
  cvt(x, h_bf, NN * DD);
  cvt(att_W1, attW1_bf, LL * DD * ATTH);
  cvt(mlp_W1, mlpW1_bf, LL * DD * DD);
  cvt(mlp_W2, mlpW2_bf, LL * DD * DD);
  cvt(pred_W, predW_bf, LL * DD * OUTD);

  mask_kernel<<<NN, 64, 0, stream>>>(x, mk_W1, mk_b1, mk_W2, mk_b2, maskbuf);

  for (int l = 0; l < LL; ++l) {
    // attention logits: hidden = relu(h @ W1 + b1) via WMMA, then dot with W2
    gemm_wmma_kernel<true, true><<<dim3(ATTH / 16, NN / 16), 32, 0, stream>>>(
        h_bf, attW1_bf + (size_t)l * DD * ATTH, att_b1 + l * ATTH, atth_bf,
        NN, ATTH, DD);
    edot_kernel<<<NN / 8, 256, 0, stream>>>(atth_bf, att_W2 + l * ATTH,
                                            att_b2 + l, e_buf);
    maxred_kernel<<<1, 1024, 0, stream>>>(e_buf, mmax);
    expp_kernel<<<NN / 256, 256, 0, stream>>>(e_buf, mmax, p_buf);

    // fused masked-softmax aggregation + degree scaling (single graph pass)
    agg_wmma_kernel<<<NN / 32, 256, 0, stream>>>(graph, p_buf, h_bf, hagg_bf);

    // GIN MLP: t1 = relu(h_agg @ W1 + b1); h2 = t1 @ W2 + b2 (f32 for BN)
    gemm_wmma_kernel<true, true><<<dim3(DD / 16, NN / 16), 32, 0, stream>>>(
        hagg_bf, mlpW1_bf + (size_t)l * DD * DD, mlp_b1 + l * DD, t1_bf,
        NN, DD, DD);
    gemm_wmma_kernel<false, false><<<dim3(DD / 16, NN / 16), 32, 0, stream>>>(
        t1_bf, mlpW2_bf + (size_t)l * DD * DD, mlp_b2 + l * DD, h2_f32,
        NN, DD, DD);

    bnstat1_kernel<<<64, 256, 0, stream>>>(h2_f32, psum, psq);
    bnstat2_kernel<<<1, 256, 0, stream>>>(psum, psq, meanb, varb);
    bnapply_kernel<<<NN, 256, 0, stream>>>(h2_f32, meanb, varb, bn_g + l * DD,
                                           bn_b + l * DD, maskbuf, l, h_bf,
                                           feat_bf);
  }

  // out = feat @ pred_W + pred_b   [8192, 64] f32
  gemm_wmma_kernel<false, false><<<dim3(OUTD / 16, NN / 16), 32, 0, stream>>>(
      feat_bf, predW_bf, pred_b, d_out, NN, OUTD, LL * DD);
}
